// attn_block_separate_add_66357244723400
// MI455X (gfx1250) — matrix operand order repaired and checked by reading, not compiled, not run on hardware
//
#include <hip/hip_runtime.h>
#include <hip/hip_bf16.h>
#include <math.h>

// ---------------------------------------------------------------------------
// Types for CDNA5 WMMA (wave32): bf16 16x16x32, f32 accumulate
// ---------------------------------------------------------------------------
typedef __bf16 v16bf __attribute__((ext_vector_type(16)));
typedef __bf16 v8bf  __attribute__((ext_vector_type(8)));
typedef float  v8f   __attribute__((ext_vector_type(8)));
typedef float  v4f   __attribute__((ext_vector_type(4)));

#define DEV __device__ __forceinline__

constexpr int Bn = 8;
constexpr int Tn = 2000;
constexpr int Dn = 512;
constexpr int Mrows = Bn * Tn;      // 16000
constexpr int TPAD  = 2016;         // padded key count for V^T tail tiles
constexpr int KPADROWS = Mrows + 64;

DEV v8f zero8() {
  v8f z;
#pragma unroll
  for (int i = 0; i < 8; ++i) z[i] = 0.0f;
  return z;
}

DEV v8f wmma_bf16(v16bf a, v16bf b, v8f c) {
  // D = A(16x32 bf16) * B(32x16 bf16) + C(16x16 f32)
  return __builtin_amdgcn_wmma_f32_16x16x32_bf16(
      /*neg_a=*/false, a, /*neg_b=*/false, b,
      /*c_mod=*/(short)0, c, /*reuse_a=*/false, /*reuse_b=*/false);
}

// A-matrix fragment (16x32, 16-bit): lane holds row M=lane%16; halves [0..7] are
// K = kbase + (lane<16?0:8) + i, halves [8..15] are K = kbase+16+(lane<16?0:8)+i.
DEV v16bf load_a_f32(const float* __restrict__ rowp, int kbase,
                     const float* __restrict__ auxp, bool useaux) {
  const int half = (threadIdx.x >> 4) & 1;
  const int c0 = kbase + half * 8;
  const int c1 = kbase + 16 + half * 8;
  v4f x0 = *(const v4f*)(rowp + c0);
  v4f x1 = *(const v4f*)(rowp + c0 + 4);
  v4f x2 = *(const v4f*)(rowp + c1);
  v4f x3 = *(const v4f*)(rowp + c1 + 4);
  if (useaux) {
    v4f a0 = *(const v4f*)(auxp + c0);
    v4f a1 = *(const v4f*)(auxp + c0 + 4);
    v4f a2 = *(const v4f*)(auxp + c1);
    v4f a3 = *(const v4f*)(auxp + c1 + 4);
    x0 += a0; x1 += a1; x2 += a2; x3 += a3;
  }
  v16bf r;
#pragma unroll
  for (int i = 0; i < 4; ++i) {
    r[i]      = (__bf16)x0[i];
    r[4 + i]  = (__bf16)x1[i];
    r[8 + i]  = (__bf16)x2[i];
    r[12 + i] = (__bf16)x3[i];
  }
  return r;
}

DEV v16bf load_a_bf16(const __bf16* __restrict__ rowp, int kbase) {
  const int half = (threadIdx.x >> 4) & 1;
  v8bf a = *(const v8bf*)(rowp + kbase + half * 8);
  v8bf b = *(const v8bf*)(rowp + kbase + 16 + half * 8);
  v16bf r;
#pragma unroll
  for (int i = 0; i < 8; ++i) { r[i] = a[i]; r[8 + i] = b[i]; }
  return r;
}

// B-matrix fragment (32x16, 16-bit): lane holds column N; halves [0..7] are
// K = kbase + half*8 + i, halves [8..15] are K = kbase+16+half*8+i. Source laid out [N][K] row-major (contiguous K).
DEV v16bf load_b_bf16(const __bf16* __restrict__ colp /* = base + N*ld */, int kbase) {
  const int half = (threadIdx.x >> 4) & 1;
  v8bf a = *(const v8bf*)(colp + kbase + half * 8);
  v8bf b = *(const v8bf*)(colp + kbase + 16 + half * 8);
  v16bf r;
#pragma unroll
  for (int i = 0; i < 8; ++i) { r[i] = a[i]; r[8 + i] = b[i]; }
  return r;
}

// ---------------------------------------------------------------------------
// Weight convert: W (f32, [K][N]) -> W^T (bf16, [N][K])
// ---------------------------------------------------------------------------
__global__ __launch_bounds__(256) void cvt_wT_k(const float* __restrict__ w,
                                                __bf16* __restrict__ wt) {
  int idx = blockIdx.x * 256 + threadIdx.x;   // 512*512 elements
  int k = idx >> 9;
  int n = idx & 511;
  wt[(size_t)n * Dn + k] = (__bf16)w[idx];
}

// ---------------------------------------------------------------------------
// GEMM: out[M x 512] = A[M x 512] @ W^T' + bias (+ aux on K or N features)
// Block = 256 threads (8 waves); block owns 16 rows x full 512 cols.
// Wave w owns cols [64w, 64w+64) with 4 f32 WMMA accumulators.
// K-loop is register double-buffered so loads for kc+32 overlap WMMAs for kc.
// ---------------------------------------------------------------------------
template <bool ABF16, bool OUTBF16, bool AUXK, bool AUXN>
__global__ __launch_bounds__(256) void gemm512_k(const void* __restrict__ A_,
                                                 const __bf16* __restrict__ BT,
                                                 const float* __restrict__ bias,
                                                 const float* __restrict__ aux,
                                                 void* __restrict__ out_) {
  const int wave = threadIdx.x >> 5;
  const int lane = threadIdx.x & 31;
  const int half = lane >> 4;
  const int n16  = lane & 15;
  const int mbase = blockIdx.x * 16;
  const int batch = mbase / Tn;           // 16-row tiles never straddle batches
  const int nbase = wave * 64;
  const int mrow  = mbase + n16;

  const float* auxp = (AUXK || AUXN) ? (aux + (size_t)batch * Dn) : nullptr;

  v8f acc0 = zero8(), acc1 = zero8(), acc2 = zero8(), acc3 = zero8();

  const __bf16* b0p = BT + (size_t)(nbase + n16) * Dn;
  const __bf16* b1p = b0p + (size_t)16 * Dn;
  const __bf16* b2p = b0p + (size_t)32 * Dn;
  const __bf16* b3p = b0p + (size_t)48 * Dn;

  auto loadA = [&](int kc) -> v16bf {
    if (ABF16)
      return load_a_bf16(((const __bf16*)A_) + (size_t)mrow * Dn, kc);
    else
      return load_a_f32(((const float*)A_) + (size_t)mrow * Dn, kc, auxp, AUXK);
  };

  // prologue fragments
  v16bf a  = loadA(0);
  v16bf b0 = load_b_bf16(b0p, 0);
  v16bf b1 = load_b_bf16(b1p, 0);
  v16bf b2 = load_b_bf16(b2p, 0);
  v16bf b3 = load_b_bf16(b3p, 0);

  for (int kc = 0; kc < Dn; kc += 32) {
    // stage next iteration's fragments (loads overlap the WMMA chain below)
    v16bf an = a, b0n = b0, b1n = b1, b2n = b2, b3n = b3;
    const int kn = kc + 32;
    if (kn < Dn) {
      an  = loadA(kn);
      b0n = load_b_bf16(b0p, kn);
      b1n = load_b_bf16(b1p, kn);
      b2n = load_b_bf16(b2p, kn);
      b3n = load_b_bf16(b3p, kn);
    }
    acc0 = wmma_bf16(a, b0, acc0);
    acc1 = wmma_bf16(a, b1, acc1);
    acc2 = wmma_bf16(a, b2, acc2);
    acc3 = wmma_bf16(a, b3, acc3);
    a = an; b0 = b0n; b1 = b1n; b2 = b2n; b3 = b3n;
  }

#pragma unroll
  for (int s = 0; s < 4; ++s) {
    v8f acc = (s == 0) ? acc0 : (s == 1) ? acc1 : (s == 2) ? acc2 : acc3;
    const int n = nbase + s * 16 + n16;
    float bb = bias[n];
    if (AUXN) bb += auxp[n];
#pragma unroll
    for (int r = 0; r < 8; ++r) {
      const int m = mbase + r + half * 8;      // C/D layout: M = reg + 8*laneHalf
      const float val = acc[r] + bb;
      if (OUTBF16)
        ((__bf16*)out_)[(size_t)m * Dn + n] = (__bf16)val;
      else
        ((float*)out_)[(size_t)m * Dn + n] = val;
    }
  }
}

// ---------------------------------------------------------------------------
// PReLU + LayerNorm over each 512-wide row; one wave per row, 8 rows per block.
// ---------------------------------------------------------------------------
__global__ __launch_bounds__(256) void prelu_ln_k(const float* __restrict__ y,
                                                  const float* __restrict__ alpha,
                                                  const float* __restrict__ g,
                                                  const float* __restrict__ be,
                                                  __bf16* __restrict__ out) {
  const int wave = threadIdx.x >> 5;
  const int lane = threadIdx.x & 31;
  const size_t row = (size_t)blockIdx.x * 8 + wave;
  const float a = alpha[0];
  const float* p = y + row * Dn;

  float x[16];
#pragma unroll
  for (int i = 0; i < 4; ++i) {
    v4f v = *(const v4f*)(p + lane * 16 + i * 4);
#pragma unroll
    for (int j = 0; j < 4; ++j) {
      float t = v[j];
      x[i * 4 + j] = (t >= 0.0f) ? t : a * t;
    }
  }
  float s = 0.0f, s2 = 0.0f;
#pragma unroll
  for (int i = 0; i < 16; ++i) { s += x[i]; s2 += x[i] * x[i]; }
#pragma unroll
  for (int off = 16; off > 0; off >>= 1) {
    s  += __shfl_xor(s, off, 32);
    s2 += __shfl_xor(s2, off, 32);
  }
  const float mean = s * (1.0f / Dn);
  const float var  = s2 * (1.0f / Dn) - mean * mean;
  const float rinv = rsqrtf(var + 1e-5f);

  __bf16* o = out + row * Dn;
#pragma unroll
  for (int i = 0; i < 16; ++i) {
    const int c = lane * 16 + i;
    o[c] = (__bf16)((x[i] - mean) * rinv * g[c] + be[c]);
  }
}

// ---------------------------------------------------------------------------
// v = sigmoid(s) * tanh(t), written transposed: vT[b][d][t] (t padded to 2016)
// ---------------------------------------------------------------------------
__global__ __launch_bounds__(256) void gate_vT_k(const float* __restrict__ s,
                                                 const float* __restrict__ t,
                                                 __bf16* __restrict__ vT) {
  const size_t idx = (size_t)blockIdx.x * 256 + threadIdx.x;   // < 16000*512
  const int d = (int)(idx & (Dn - 1));
  const size_t row = idx >> 9;
  const int b  = (int)(row / Tn);
  const int tt = (int)(row % Tn);
  const float sv = s[idx];
  const float tv = t[idx];
  const float val = (1.0f / (1.0f + __expf(-sv))) * tanhf(tv);
  vT[((size_t)b * Dn + d) * TPAD + tt] = (__bf16)val;
}

// ---------------------------------------------------------------------------
// Flash attention with the reference's exact masking semantics.
// Grid: Bn * 125 blocks; block = 128 threads (4 waves) owning 16 query rows.
// Each iteration handles a 32-key tile:
//   S: each wave computes its 128-wide K-slice partial (8 WMMAs) -> ds_add_f32
//   softmax: 16 threads, online (m, l) update, P -> LDS bf16 (A layout source)
//   O: per-row rescale + P@V (wave owns 128 output cols, 8 WMMAs)
// B fragments are batch-loaded before each WMMA chain; next tile is prefetched.
// ---------------------------------------------------------------------------
__global__ __launch_bounds__(128) void flash_attn_k(const __bf16* __restrict__ q,
                                                    const __bf16* __restrict__ kmat,
                                                    const __bf16* __restrict__ vT,
                                                    const int* __restrict__ input_len,
                                                    float* __restrict__ out) {
  __shared__ __align__(16) float  Slds[16][32];
  __shared__ __align__(16) __bf16 Plds[16][32];
  __shared__ float MrowS[16], LrowS[16], ScaleS[16];

  const int wave = threadIdx.x >> 5;
  const int lane = threadIdx.x & 31;
  const int half = lane >> 4;
  const int n16  = lane & 15;

  const int b     = blockIdx.x / (Tn / 16);
  const int mtile = blockIdx.x % (Tn / 16);
  const int mbase = mtile * 16;
  const size_t qrow0 = (size_t)b * Tn + mbase;
  const int L = input_len[b];
  const float invsq = 0.044194173824159216f;   // 1/sqrt(512)
  const float NEGINF = -__builtin_inff();

  // Preload Q A-fragments for this wave's 128-wide K slice.
  v16bf qa[4];
#pragma unroll
  for (int i = 0; i < 4; ++i)
    qa[i] = load_a_bf16(q + (qrow0 + n16) * Dn, wave * 128 + i * 32);

  v8f oacc[8];
#pragma unroll
  for (int s = 0; s < 8; ++s) oacc[s] = zero8();

  if (threadIdx.x < 16) { MrowS[threadIdx.x] = NEGINF; LrowS[threadIdx.x] = 0.0f; }
  __syncthreads();

  const int NST = (Tn + 31) / 32;   // 63 (tail tile masked)
  for (int js = 0; js < NST; ++js) {
    const int sbase = js * 32;

    // prefetch next tile: one key row and one V^T column segment per lane
    if (js + 1 < NST) {
      const int nb = sbase + 32;
      __builtin_prefetch(kmat + ((size_t)b * Tn + nb + lane) * Dn + wave * 128, 0, 1);
      __builtin_prefetch(vT + ((size_t)b * Dn + wave * 128 + lane) * TPAD + nb, 0, 1);
    }

    // zero the shared score tile
    {
      float* sp = &Slds[0][0];
      sp[threadIdx.x] = 0.0f;
      sp[threadIdx.x + 128] = 0.0f;
      sp[threadIdx.x + 256] = 0.0f;
      sp[threadIdx.x + 384] = 0.0f;
    }
    __syncthreads();

    // partial S = Q(16x128slice) @ K_tile^T : batch all loads, then WMMA chain
#pragma unroll
    for (int sub = 0; sub < 2; ++sub) {
      const int krow = sbase + sub * 16 + n16;                 // key row (padded buf)
      const __bf16* kp = kmat + ((size_t)b * Tn + krow) * Dn;  // contiguous features
      v16bf kf0 = load_b_bf16(kp, wave * 128);
      v16bf kf1 = load_b_bf16(kp, wave * 128 + 32);
      v16bf kf2 = load_b_bf16(kp, wave * 128 + 64);
      v16bf kf3 = load_b_bf16(kp, wave * 128 + 96);
      v8f acc = zero8();
      acc = wmma_bf16(qa[0], kf0, acc);
      acc = wmma_bf16(qa[1], kf1, acc);
      acc = wmma_bf16(qa[2], kf2, acc);
      acc = wmma_bf16(qa[3], kf3, acc);
#pragma unroll
      for (int r = 0; r < 8; ++r)
        atomicAdd(&Slds[r + half * 8][sub * 16 + n16], acc[r]);
    }
    __syncthreads();

    // online softmax with reference masking, by 16 lanes (one per query row)
    if (threadIdx.x < 16) {
      const int tid = threadIdx.x;
      const int grow = mbase + tid;
      const float mo = MrowS[tid];
      const float lo = LrowS[tid];
      float vals[32];
      float tmax = NEGINF;
#pragma unroll
      for (int j = 0; j < 32; ++j) {
        const int col = sbase + j;
        float v = Slds[tid][j] * invsq;
        v = (col <= grow) ? v : 0.0f;          // causal multiply-mask
        v = (v == 0.0f) ? NEGINF : v;          // reference's w==0 -> -inf quirk
        if (grow < L && col >= L) v = NEGINF;  // drop mask
        if (grow >= L) v = 0.0f;               // rows past length: uniform
        if (col >= Tn) v = NEGINF;             // tail padding
        vals[j] = v;
        tmax = fmaxf(tmax, v);
      }
      const float mn = fmaxf(mo, tmax);
      float sc, psum = 0.0f;
      if (mn == NEGINF) {
        sc = 1.0f;
#pragma unroll
        for (int j = 0; j < 32; ++j) Plds[tid][j] = (__bf16)0.0f;
      } else {
        sc = __expf(mo - mn);
#pragma unroll
        for (int j = 0; j < 32; ++j) {
          const float pv = __expf(vals[j] - mn);
          psum += pv;
          Plds[tid][j] = (__bf16)pv;
        }
      }
      MrowS[tid]  = mn;
      LrowS[tid]  = lo * sc + psum;
      ScaleS[tid] = sc;
    }
    __syncthreads();

    // rescale running O and accumulate P @ V_tile for this wave's 128 cols
    float scr[8];
#pragma unroll
    for (int r = 0; r < 8; ++r) scr[r] = ScaleS[half * 8 + r];
#pragma unroll
    for (int s = 0; s < 8; ++s)
#pragma unroll
      for (int r = 0; r < 8; ++r) oacc[s][r] *= scr[r];

    v16bf pa = load_a_bf16(&Plds[n16][0], 0);   // P row is 32 contiguous bf16
#pragma unroll
    for (int g = 0; g < 2; ++g) {
      v16bf vf[4];
#pragma unroll
      for (int s = 0; s < 4; ++s) {
        const int n = wave * 128 + (g * 4 + s) * 16 + n16;
        vf[s] = load_b_bf16(vT + ((size_t)b * Dn + n) * TPAD + sbase, 0);
      }
#pragma unroll
      for (int s = 0; s < 4; ++s)
        oacc[g * 4 + s] = wmma_bf16(pa, vf[s], oacc[g * 4 + s]);
    }
    __syncthreads();   // protect Plds/ScaleS/Slds before next iteration
  }

  // finalize: O / l
  float li[8];
#pragma unroll
  for (int r = 0; r < 8; ++r) {
    const float l = LrowS[half * 8 + r];
    li[r] = (l != 0.0f) ? (1.0f / l) : 0.0f;
  }
#pragma unroll
  for (int s = 0; s < 8; ++s) {
    const int n = wave * 128 + s * 16 + n16;
#pragma unroll
    for (int r = 0; r < 8; ++r) {
      const int m = mbase + r + half * 8;
      out[((size_t)b * Tn + m) * Dn + n] = oacc[s][r] * li[r];
    }
  }
}

// ---------------------------------------------------------------------------
// Host orchestration
// ---------------------------------------------------------------------------
extern "C" void kernel_launch(void* const* d_in, const int* in_sizes, int n_in,
                              void* d_out, int out_size, void* d_ws, size_t ws_size,
                              hipStream_t stream) {
  (void)in_sizes; (void)n_in; (void)out_size; (void)ws_size;

  const float* query   = (const float*)d_in[0];
  const float* value   = (const float*)d_in[1];
  const float* aux     = (const float*)d_in[2];
  const int*   ilen    = (const int*)d_in[3];
  const float* q_lin_w = (const float*)d_in[4];
  const float* q_lin_b = (const float*)d_in[5];
  const float* qv_w    = (const float*)d_in[6];
  const float* qv_b    = (const float*)d_in[7];
  const float* qv_a    = (const float*)d_in[8];
  const float* qv_g    = (const float*)d_in[9];
  const float* qv_be   = (const float*)d_in[10];
  const float* kv_w    = (const float*)d_in[11];
  const float* kv_b    = (const float*)d_in[12];
  const float* kv_a    = (const float*)d_in[13];
  const float* kv_g    = (const float*)d_in[14];
  const float* kv_be   = (const float*)d_in[15];
  const float* vv_w    = (const float*)d_in[16];
  const float* vv_b    = (const float*)d_in[17];
  const float* vv_a    = (const float*)d_in[18];
  const float* vv_g    = (const float*)d_in[19];
  const float* vv_be   = (const float*)d_in[20];
  const float* sig_w   = (const float*)d_in[21];
  const float* sig_b   = (const float*)d_in[22];
  const float* tan_w   = (const float*)d_in[23];
  const float* tan_b   = (const float*)d_in[24];

  char* wsp = (char*)d_ws;
  auto alloc = [&](size_t bytes) -> char* {
    char* p = wsp;
    wsp += (bytes + 255) & ~(size_t)255;
    return p;
  };

  __bf16* WT[6];
  for (int i = 0; i < 6; ++i) WT[i] = (__bf16*)alloc((size_t)Dn * Dn * 2);
  __bf16* qpre = (__bf16*)alloc((size_t)Mrows * Dn * 2);        // also reused as vmid
  __bf16* qbf  = (__bf16*)alloc((size_t)Mrows * Dn * 2);
  __bf16* kbf  = (__bf16*)alloc((size_t)KPADROWS * Dn * 2);     // padded tail rows
  __bf16* vTb  = (__bf16*)alloc((size_t)Bn * Dn * TPAD * 2);
  float*  ybuf = (float*)alloc((size_t)Mrows * Dn * 4);
  float*  tbuf = (float*)alloc((size_t)Mrows * Dn * 4);

  // 1) bf16 transposed weights
  const float* Wsrc[6] = {q_lin_w, qv_w, kv_w, vv_w, sig_w, tan_w};
  for (int i = 0; i < 6; ++i)
    cvt_wT_k<<<(Dn * Dn) / 256, 256, 0, stream>>>(Wsrc[i], WT[i]);

  const int GB = Mrows / 16;   // 1000 GEMM blocks

  // 2) q_pre = query @ Wq + b + aux  -> bf16
  gemm512_k<false, true, false, true><<<GB, 256, 0, stream>>>(query, WT[0], q_lin_b, aux, qpre);
  // 3) q = LN(PReLU(q_pre @ Wqv + b))
  gemm512_k<true, false, false, false><<<GB, 256, 0, stream>>>(qpre, WT[1], qv_b, nullptr, ybuf);
  prelu_ln_k<<<Mrows / 8, 256, 0, stream>>>(ybuf, qv_a, qv_g, qv_be, qbf);
  // 4) k = LN(PReLU((value+aux) @ Wkv + b))
  gemm512_k<false, false, true, false><<<GB, 256, 0, stream>>>(value, WT[2], kv_b, aux, ybuf);
  prelu_ln_k<<<Mrows / 8, 256, 0, stream>>>(ybuf, kv_a, kv_g, kv_be, kbf);
  // 5) vmid = LN(PReLU((value+aux) @ Wvv + b))  (reuse qpre slot)
  gemm512_k<false, false, true, false><<<GB, 256, 0, stream>>>(value, WT[3], vv_b, aux, ybuf);
  prelu_ln_k<<<Mrows / 8, 256, 0, stream>>>(ybuf, vv_a, vv_g, vv_be, qpre);
  // 6) v = sigmoid(vmid @ Ws + bs) * tanh(vmid @ Wt + bt), stored transposed bf16
  gemm512_k<true, false, false, false><<<GB, 256, 0, stream>>>(qpre, WT[4], sig_b, nullptr, ybuf);
  gemm512_k<true, false, false, false><<<GB, 256, 0, stream>>>(qpre, WT[5], tan_b, nullptr, tbuf);
  gate_vT_k<<<((size_t)Mrows * Dn) / 256, 256, 0, stream>>>(ybuf, tbuf, vTb);
  // 7) masked softmax attention
  flash_attn_k<<<Bn * (Tn / 16), 128, 0, stream>>>(qbf, kbf, vTb, ilen, (float*)d_out);
}
